// PointTransformerLayer_81097572483176
// MI455X (gfx1250) — matrix operand order repaired and checked by reading, not compiled, not run on hardware
//
#include <hip/hip_runtime.h>
#include <hip/hip_bf16.h>

typedef _Float16 f16;
typedef f16  v16h __attribute__((ext_vector_type(16)));
typedef f16  v8h  __attribute__((ext_vector_type(8)));
typedef float v8f __attribute__((ext_vector_type(8)));

#define DIMC 64
#define NPTS 128
#define BTOT 32
#define PIN  34
#define QKVW 192

// workspace layout (floats unless noted)
#define QKV_ELEMS   (BTOT * NPTS * QKVW)          // 786432 f32
#define P_ELEMS     (BTOT * NPTS * PIN)           // 139264 f32
#define WF_BYTE_OFF ((size_t)(QKV_ELEMS + P_ELEMS) * sizeof(float))
// swizzled f16 weight fragments: w1(4096) w2(4096) w3(16384) w4(16384) halves
#define WF_HALVES   40960

static __device__ __forceinline__ v8f wmma_f16(v16h a, v16h b, v8f c) {
  return __builtin_amdgcn_wmma_f32_16x16x32_f16(false, a, false, b, (short)0, c,
                                                false, false);
}

// ---------------------------------------------------------------------------
// Kernel A: build p (permuted coords), qkv = x @ w_qkv.T, and swizzle all MLP
// weights into wave32 WMMA B-fragment layout (f16) in the workspace.
// ---------------------------------------------------------------------------
__global__ __launch_bounds__(256) void pt_precompute(
    const float* __restrict__ x, const float* __restrict__ pos,
    const float* __restrict__ w_qkv,
    const float* __restrict__ pm_w1, const float* __restrict__ pm_w2,
    const float* __restrict__ am_w1, const float* __restrict__ am_w2,
    float* __restrict__ qkv_ws, float* __restrict__ p_ws,
    f16* __restrict__ wf) {
  const int bt = blockIdx.x;
  const int tid = threadIdx.x;

  // p[bt][n][f] with f = c*17 + j  <-  pos[b][t][j][c][n]
  for (int idx = tid; idx < NPTS * PIN; idx += 256) {
    int n = idx / PIN, f = idx % PIN;
    int c = f / 17, j = f % 17;
    p_ws[(bt * NPTS + n) * PIN + f] = pos[((bt * 17 + j) * 3 + c) * NPTS + n];
  }

  // qkv[bt][row][col] = sum_k x[bt][row][k] * w_qkv[col][k]
  for (int o = tid; o < NPTS * QKVW; o += 256) {
    int row = o / QKVW, col = o % QKVW;
    const float* xr = x + (bt * NPTS + row) * DIMC;
    const float* wr = w_qkv + col * DIMC;
    float s = 0.f;
#pragma unroll 8
    for (int k = 0; k < DIMC; ++k) s += xr[k] * wr[k];
    qkv_ws[(bt * NPTS + row) * QKVW + col] = s;
  }

  if (bt == 0) {
    // B-fragment layout used here (contiguous k order, same as the A loads):
    //   element e of lane l holds B[K][N] with K = e + 16*(l>>4), N = l&15
    auto swz = [&](f16* dst, const float* src, int Ksrc, int nT, int nKC) {
      int total = nKC * nT * 512;
      for (int idx = tid; idx < total; idx += 256) {
        int e = idx & 15;
        int l = (idx >> 4) & 31;
        int r = idx >> 9;
        int t = r % nT;
        int kc = r / nT;
        int K = kc * 32 + e + ((l >> 4) << 4);
        int N = t * 16 + (l & 15);
        float v = (K < Ksrc) ? src[N * Ksrc + K] : 0.f;
        dst[idx] = (f16)v;
      }
    };
    swz(wf,          pm_w1, PIN, 4,  2);  // B1: [64x64], K padded from 34
    swz(wf + 4096,   pm_w2, 64,  4,  2);  // B2: [64x64]
    swz(wf + 8192,   am_w1, 64,  16, 2);  // B3: [64x256]
    swz(wf + 24576,  am_w2, 256, 4,  8);  // B4: [256x64]
  }
}

// ---------------------------------------------------------------------------
// Kernel B: fully fused point-transformer layer. One block per (bt, i).
// 8 waves; wave w owns key rows j in [16w, 16w+16). Weight fragments are
// copied to LDS once per block; all B-matrix reads come from LDS.
// ---------------------------------------------------------------------------
__global__ __launch_bounds__(256) void pt_attn(
    const float* __restrict__ qkv_ws, const float* __restrict__ p_ws,
    const f16* __restrict__ wf,
    const float* __restrict__ b1g, const float* __restrict__ b2g,
    const float* __restrict__ b3g, const float* __restrict__ b4g,
    float* __restrict__ out) {
  __shared__ __align__(16) f16 wfL[WF_HALVES];   // 80 KB swizzled weights
  __shared__ __align__(16) f16 stage[8 * 16 * 72];  // per-wave [16x72] bounce
  __shared__ float pmax[8 * 64], psum[8 * 64], pacc[8 * 64];

  const int tid = threadIdx.x;
  const int wv = tid >> 5;
  const int ln = tid & 31;
  const int hh = ln >> 4;     // lane half (0/1)
  const int cn = ln & 15;     // column within tile / row for A-frag reads
  const int bt = blockIdx.x >> 7;
  const int qi = blockIdx.x & 127;
  const int jb = wv * 16;

  // ---- cooperative copy of all weight fragments into LDS -----------------
  {
    const uint4* s4 = (const uint4*)wf;
    uint4* d4 = (uint4*)wfL;
#pragma unroll 4
    for (int i = tid; i < WF_HALVES / 8; i += 256) d4[i] = s4[i];
  }

  f16* st = &stage[wv * 16 * 72];
  const float* pbt = p_ws + bt * NPTS * PIN;
  const float* qkbt = qkv_ws + (size_t)bt * NPTS * QKVW;

  // A-fragment (16x32 f16): element e of lane l holds A[M][K] with
  //   M = l&15, K = kc*32 + e + 16*(l>>4)
  // From a row-major [16 x 72] f16 stage this is two contiguous v8h loads.
  auto loadA = [&](int kc) -> v16h {
    union { v16h v; v8h h[2]; } u;
    const f16* base = &st[cn * 72 + kc * 32 + (hh << 4)];
    u.h[0] = *(const v8h*)(base);
    u.h[1] = *(const v8h*)(base + 8);
    return u.v;
  };
  auto loadB = [&](const f16* arr, int frag) -> v16h {
    return *(const v16h*)(arr + frag * 512 + ln * 16);
  };

  // ---- stage rel[j][f] = p[i][f] - p[j][f], zero-padded to 64 cols --------
#pragma unroll 4
  for (int it = 0; it < 32; ++it) {
    int flat = ln + 32 * it;       // 16*64 elems per wave
    int row = flat >> 6;
    int col = flat & 63;
    float v = 0.f;
    if (col < PIN) v = pbt[qi * PIN + col] - pbt[(jb + row) * PIN + col];
    st[row * 72 + col] = (f16)v;
  }

  __syncthreads();   // weights visible to all waves

  v16h a0 = loadA(0), a1 = loadA(1);

  // ---- GEMM1: relu(rel @ W1 + b1) -> stage -------------------------------
  const f16* W1 = wfL;
#pragma unroll
  for (int t = 0; t < 4; ++t) {
    v8f c = {0.f, 0.f, 0.f, 0.f, 0.f, 0.f, 0.f, 0.f};
    c = wmma_f16(a0, loadB(W1, t), c);
    c = wmma_f16(a1, loadB(W1, 4 + t), c);
    float bias = b1g[t * 16 + cn];
#pragma unroll
    for (int r = 0; r < 8; ++r) {
      float v = c[r] + bias;
      st[(r + 8 * hh) * 72 + t * 16 + cn] = (f16)(v > 0.f ? v : 0.f);
    }
  }

  // ---- GEMM2: rpe = act @ W2 + b2 (kept in registers) --------------------
  a0 = loadA(0); a1 = loadA(1);
  const f16* W2 = wfL + 4096;
  v8f rpe[4];
#pragma unroll
  for (int t = 0; t < 4; ++t) {
    v8f c = {0.f, 0.f, 0.f, 0.f, 0.f, 0.f, 0.f, 0.f};
    c = wmma_f16(a0, loadB(W2, t), c);
    c = wmma_f16(a1, loadB(W2, 4 + t), c);
    float bias = b2g[t * 16 + cn];
#pragma unroll
    for (int r = 0; r < 8; ++r) rpe[t][r] = c[r] + bias;
  }

  // ---- h = rpe + q[i] - k[j] -> stage ------------------------------------
#pragma unroll
  for (int t = 0; t < 4; ++t) {
    int d = t * 16 + cn;
    float qv = qkbt[qi * QKVW + d];
#pragma unroll
    for (int r = 0; r < 8; ++r) {
      int jg = jb + r + 8 * hh;
      float h = rpe[t][r] + qv - qkbt[jg * QKVW + 64 + d];
      st[(r + 8 * hh) * 72 + d] = (f16)h;
    }
  }
  a0 = loadA(0); a1 = loadA(1);

  // ---- GEMM3 + GEMM4 fused over hidden chunks of 32 ----------------------
  const f16* W3 = wfL + 8192;
  const f16* W4 = wfL + 24576;
  v8f sim[4];
#pragma unroll
  for (int t = 0; t < 4; ++t)
    sim[t] = (v8f){0.f, 0.f, 0.f, 0.f, 0.f, 0.f, 0.f, 0.f};

#pragma unroll
  for (int kc = 0; kc < 8; ++kc) {
#pragma unroll
    for (int u = 0; u < 2; ++u) {
      int t2 = kc * 2 + u;
      v8f c = {0.f, 0.f, 0.f, 0.f, 0.f, 0.f, 0.f, 0.f};
      c = wmma_f16(a0, loadB(W3, t2), c);
      c = wmma_f16(a1, loadB(W3, 16 + t2), c);
      float bias = b3g[t2 * 16 + cn];
#pragma unroll
      for (int r = 0; r < 8; ++r) {
        float v = c[r] + bias;
        st[(r + 8 * hh) * 72 + u * 16 + cn] = (f16)(v > 0.f ? v : 0.f);
      }
    }
    union { v16h v; v8h h[2]; } u4;
    const f16* base = &st[cn * 72 + (hh << 4)];
    u4.h[0] = *(const v8h*)(base);
    u4.h[1] = *(const v8h*)(base + 8);
#pragma unroll
    for (int t = 0; t < 4; ++t)
      sim[t] = wmma_f16(u4.v, loadB(W4, kc * 4 + t), sim[t]);
  }

  // ---- softmax over j (per column d) + weighted sum of vpair -------------
#pragma unroll
  for (int t = 0; t < 4; ++t) {
    int d = t * 16 + cn;
    float bias = b4g[d];
    float vp[8];
#pragma unroll
    for (int r = 0; r < 8; ++r) {
      sim[t][r] += bias;
      int jg = jb + r + 8 * hh;
      vp[r] = rpe[t][r] + qkbt[jg * QKVW + 128 + d];
    }
    float m = sim[t][0];
#pragma unroll
    for (int r = 1; r < 8; ++r) m = fmaxf(m, sim[t][r]);
    m = fmaxf(m, __shfl_xor(m, 16, 32));   // combine lane halves (rows 0-7 | 8-15)
    float s = 0.f, a = 0.f;
#pragma unroll
    for (int r = 0; r < 8; ++r) {
      float e = __expf(sim[t][r] - m);
      s += e;
      a += e * vp[r];
    }
    s += __shfl_xor(s, 16, 32);
    a += __shfl_xor(a, 16, 32);
    if (hh == 0) {
      pmax[wv * 64 + d] = m;
      psum[wv * 64 + d] = s;
      pacc[wv * 64 + d] = a;
    }
  }
  __syncthreads();

  if (tid < 64) {
    int d = tid;
    float m = pmax[d];
#pragma unroll
    for (int w = 1; w < 8; ++w) m = fmaxf(m, pmax[w * 64 + d]);
    float Z = 0.f, A = 0.f;
#pragma unroll
    for (int w = 0; w < 8; ++w) {
      float f = __expf(pmax[w * 64 + d] - m);
      Z += psum[w * 64 + d] * f;
      A += pacc[w * 64 + d] * f;
    }
    out[((size_t)bt * NPTS + qi) * DIMC + d] = A / Z;
  }
}

// ---------------------------------------------------------------------------
extern "C" void kernel_launch(void* const* d_in, const int* in_sizes, int n_in,
                              void* d_out, int out_size, void* d_ws,
                              size_t ws_size, hipStream_t stream) {
  (void)in_sizes; (void)n_in; (void)out_size; (void)ws_size;
  const float* x     = (const float*)d_in[0];
  const float* pos   = (const float*)d_in[1];
  const float* w_qkv = (const float*)d_in[2];
  const float* pm_w1 = (const float*)d_in[3];
  const float* pm_b1 = (const float*)d_in[4];
  const float* pm_w2 = (const float*)d_in[5];
  const float* pm_b2 = (const float*)d_in[6];
  const float* am_w1 = (const float*)d_in[7];
  const float* am_b1 = (const float*)d_in[8];
  const float* am_w2 = (const float*)d_in[9];
  const float* am_b2 = (const float*)d_in[10];

  float* qkv_ws = (float*)d_ws;
  float* p_ws   = qkv_ws + QKV_ELEMS;
  f16*   wf     = (f16*)((char*)d_ws + WF_BYTE_OFF);

  pt_precompute<<<BTOT, 256, 0, stream>>>(x, pos, w_qkv, pm_w1, pm_w2, am_w1,
                                          am_w2, qkv_ws, p_ws, wf);
  pt_attn<<<BTOT * NPTS, 256, 0, stream>>>(qkv_ws, p_ws, wf, pm_b1, pm_b2,
                                           am_b1, am_b2, (float*)d_out);
}
